// NAMModel_60112362275084
// MI455X (gfx1250) — matrix operand order repaired and checked by reading, not compiled, not run on hardware
//
#include <hip/hip_runtime.h>

typedef __attribute__((ext_vector_type(16))) _Float16 v16h;
typedef __attribute__((ext_vector_type(8)))  float    v8f;

constexpr int NF = 128;   // features
constexpr int H1 = 64;
constexpr int H2 = 64;
constexpr int H3 = 32;

union AFrag {
    v16h v;
    _Float16 h[16];
    unsigned u[8];
};

__global__ __launch_bounds__(256) void nam_init_kernel(const float* __restrict__ bias,
                                                       float* __restrict__ out, int n) {
    int i = blockIdx.x * 256 + threadIdx.x;
    if (i < n) out[i] = bias[0];
}

__global__ __launch_bounds__(256) void nam_wmma_kernel(
    const float* __restrict__ x,
    const float* __restrict__ W0, const float* __restrict__ b0,
    const float* __restrict__ W1, const float* __restrict__ b1,
    const float* __restrict__ W2, const float* __restrict__ b2,
    const float* __restrict__ W3, const float* __restrict__ b3,
    float* __restrict__ out, int N)
{
    // LDS: weights transposed to [outCol][k] in f16 so B-fragments are contiguous.
    __shared__ alignas(32) _Float16 W1t[H2][H1];          // 8 KB
    __shared__ alignas(32) _Float16 W2t[H3][H2];          // 4 KB
    __shared__ alignas(32) _Float16 stage[8][16][H1];     // 16 KB: per-wave h1 tile [M][K]
    __shared__ float W0s[H1], b0s[H1], b1s[H2], b2s[H3], W3s[H3];
    __shared__ float b3s;

    const int f   = blockIdx.x;
    const int tid = threadIdx.x;

    // ---- cooperative weight staging (f32 -> f16, transposed) ----
    {
        const float* W1f = W1 + f * H1 * H2;   // [k][n]
        for (int i = tid; i < H1 * H2; i += 256) {
            int k = i / H2, n = i % H2;
            W1t[n][k] = (_Float16)W1f[i];
        }
        const float* W2f = W2 + f * H2 * H3;   // [k][n]
        for (int i = tid; i < H2 * H3; i += 256) {
            int k = i / H3, n = i % H3;
            W2t[n][k] = (_Float16)W2f[i];
        }
        if (tid < H1) {
            W0s[tid] = W0[f * H1 + tid];
            b0s[tid] = b0[f * H1 + tid];
            b1s[tid] = b1[f * H2 + tid];
        }
        if (tid < H3) {
            b2s[tid] = b2[f * H3 + tid];
            W3s[tid] = W3[f * H3 + tid];
        }
        if (tid == 0) b3s = b3[f];
    }
    __syncthreads();

    const int lane  = tid & 31;
    const int wave  = tid >> 5;
    const int laneM = lane & 15;
    const bool hiH  = lane >= 16;
    const int kOffA = hiH ? 16 : 0;   // A-fragment K offset for high half-wave
    const int kOffB = hiH ? 16 : 0;   // B-fragment K offset for high half-wave

    const int rowsPerBlock  = N / gridDim.y;          // 1024 for N=8192, grid.y=8
    const int rowBase       = blockIdx.y * rowsPerBlock;
    const int tilesPerBlock = rowsPerBlock / 16;

    for (int t = wave; t < tilesPerBlock; t += 8) {
        const int n0 = rowBase + t * 16;

        // prefetch next tile's x values for this wave (uniform condition)
        if (t + 8 < tilesPerBlock)
            __builtin_prefetch(&x[(size_t)(n0 + 128 + laneM) * NF + f], 0, 3);

        // ---- layer 0: h0 = relu(x*W0 + b0), built directly in A-fragment layout ----
        const float xv = x[(size_t)(n0 + laneM) * NF + f];
        AFrag A0, A1;   // K = 0..31 and K = 32..63
#pragma unroll
        for (int v = 0; v < 8; ++v) {
#pragma unroll
            for (int p = 0; p < 2; ++p) {
                const int k = (2 * v + p) + kOffA;
                A0.h[2 * v + p] = (_Float16)fmaxf(fmaf(xv, W0s[k],      b0s[k]),      0.0f);
                A1.h[2 * v + p] = (_Float16)fmaxf(fmaf(xv, W0s[k + 32], b0s[k + 32]), 0.0f);
            }
        }

        // ---- layer 1: h1 = relu(h0 @ W1 + b1); 4 column tiles, K=64 in 2 steps ----
#pragma unroll
        for (int nt = 0; nt < 4; ++nt) {
            const int ncol = nt * 16 + laneM;
            const v16h B0 = *(const v16h*)&W1t[ncol][0  + kOffB];
            const v16h B1 = *(const v16h*)&W1t[ncol][32 + kOffB];
            v8f acc = {};
            acc = __builtin_amdgcn_wmma_f32_16x16x32_f16(false, A0.v, false, B0,
                                                         (short)0, acc, false, false);
            acc = __builtin_amdgcn_wmma_f32_16x16x32_f16(false, A1.v, false, B1,
                                                         (short)0, acc, false, false);
            const float bn = b1s[ncol];
#pragma unroll
            for (int r = 0; r < 8; ++r) {
                const int M = r + (hiH ? 8 : 0);
                stage[wave][M][ncol] = (_Float16)fmaxf(acc[r] + bn, 0.0f);
            }
        }

        // ---- load layer-2 A fragments from staged h1 (paired-K dword loads) ----
        AFrag C0, C1;
#pragma unroll
        for (int v = 0; v < 8; ++v) {
            const int kp = (2 * v) + kOffA;
            C0.u[v] = *(const unsigned*)&stage[wave][laneM][kp];
            C1.u[v] = *(const unsigned*)&stage[wave][laneM][kp + 32];
        }

        // ---- layer 2 + layer 3 fused: contrib = sum_j relu(h1@W2 + b2)_j * W3_j ----
        float partial[8] = {};
#pragma unroll
        for (int nt = 0; nt < 2; ++nt) {
            const int ncol = nt * 16 + laneM;
            const v16h B0 = *(const v16h*)&W2t[ncol][0  + kOffB];
            const v16h B1 = *(const v16h*)&W2t[ncol][32 + kOffB];
            v8f acc = {};
            acc = __builtin_amdgcn_wmma_f32_16x16x32_f16(false, C0.v, false, B0,
                                                         (short)0, acc, false, false);
            acc = __builtin_amdgcn_wmma_f32_16x16x32_f16(false, C1.v, false, B1,
                                                         (short)0, acc, false, false);
            const float bn = b2s[ncol];
            const float w3 = W3s[ncol];
#pragma unroll
            for (int r = 0; r < 8; ++r)
                partial[r] += fmaxf(acc[r] + bn, 0.0f) * w3;
        }

        // reduce over the 16 lanes holding the 16 columns (xor stays in half-wave)
#pragma unroll
        for (int r = 0; r < 8; ++r) {
            float s = partial[r];
            s += __shfl_xor(s, 1);
            s += __shfl_xor(s, 2);
            s += __shfl_xor(s, 4);
            s += __shfl_xor(s, 8);
            partial[r] = s + b3s;
        }

        // lanes 0 (rows 0..7) and 16 (rows 8..15) accumulate into out[n]
        if (lane == 0 || lane == 16) {
            const int mofs = hiH ? 8 : 0;
#pragma unroll
            for (int r = 0; r < 8; ++r)
                atomicAdd(&out[n0 + mofs + r], partial[r]);
        }
    }
}

extern "C" void kernel_launch(void* const* d_in, const int* in_sizes, int n_in,
                              void* d_out, int out_size, void* d_ws, size_t ws_size,
                              hipStream_t stream) {
    const float* x    = (const float*)d_in[0];
    const float* W0   = (const float*)d_in[1];
    const float* b0   = (const float*)d_in[2];
    const float* W1   = (const float*)d_in[3];
    const float* b1   = (const float*)d_in[4];
    const float* W2   = (const float*)d_in[5];
    const float* b2   = (const float*)d_in[6];
    const float* W3   = (const float*)d_in[7];
    const float* b3   = (const float*)d_in[8];
    const float* bias = (const float*)d_in[9];
    float* out = (float*)d_out;

    const int N = in_sizes[0] / NF;   // 8192

    nam_init_kernel<<<(N + 255) / 256, 256, 0, stream>>>(bias, out, N);

    dim3 grid(NF, 8);                 // 128 features x 8 row-chunks = 1024 blocks
    nam_wmma_kernel<<<grid, 256, 0, stream>>>(x, W0, b0, W1, b1, W2, b2, W3, b3, out, N);
}
